// SpectralProcessingModel_58317065945615
// MI455X (gfx1250) — compile-verified, matrix operand order checked by reading, not run on hardware
//
#include <hip/hip_runtime.h>
#include <hip/hip_bf16.h>

// ---------------------------------------------------------------------------
// Types for CDNA5 WMMA (wave32): v_wmma_f32_16x16x32_bf16
// ---------------------------------------------------------------------------
typedef __attribute__((ext_vector_type(16))) __bf16 v16bf;
typedef __attribute__((ext_vector_type(8)))  float  v8f;

union Frag { unsigned int u[8]; v16bf v; };

// 16-bit A/B operand K-offset for (VGPR v, lane l): pairs (k0,k0+1)
__device__ __forceinline__ int akoff(int v, int lane) {
  return ((v >> 2) << 4) + ((lane >> 4) << 3) + ((v & 3) << 1);
}

__device__ __forceinline__ v8f wmma_bf16(const Frag& a, const Frag& b, v8f c) {
  return __builtin_amdgcn_wmma_f32_16x16x32_bf16(
      false, a.v, false, b.v, (short)0, c, false, false);
}

// CDNA5 async global->LDS DMA (ASYNCcnt-tracked), per-lane b128.
__device__ __forceinline__ void async_copy_b128(unsigned lds_off, const void* gaddr) {
  asm volatile("global_load_async_to_lds_b128 %0, %1, off"
               :: "v"(lds_off), "v"(gaddr) : "memory");
}
__device__ __forceinline__ void wait_async_le3() {
  asm volatile("s_wait_asynccnt 0x3" ::: "memory");
}
__device__ __forceinline__ void wait_async_0() {
  asm volatile("s_wait_asynccnt 0x0" ::: "memory");
}

// ---------------------------------------------------------------------------
// Model dims
// ---------------------------------------------------------------------------
#define BATCH   2
#define CHN     16
#define NTOK    2112        // 64 * 33
#define DMODEL  1024
#define NHEADS  8
#define HDIM    128
#define NLAYERS 8
#define MROWS   (BATCH * NTOK)   // 4224 = 33 * 128
#define HEADSEG ((size_t)BATCH * CHN * 1024 * NTOK)  // 69,206,016

// ---------------------------------------------------------------------------
// f32 -> bf16 conversion
// ---------------------------------------------------------------------------
__global__ void to_bf16_kernel(const float* __restrict__ s, __bf16* __restrict__ d, size_t n) {
  size_t i = (size_t)blockIdx.x * blockDim.x + threadIdx.x;
  if (i < n) d[i] = (__bf16)s[i];
}

// ---------------------------------------------------------------------------
// FFT: rfft2(norm='ortho') + fftshift(axis=2) + row reorder -> feat (B,NTOK,32)
// ---------------------------------------------------------------------------
__device__ __forceinline__ int row_order_dev(int p) {
  int i = (p + 1) >> 1;
  int r;
  if ((p & 1) && (32 + i) < 64) r = 32 + i; else r = 32 - i;
  return r;
}

__global__ void fft_feat_kernel(const float* __restrict__ x, float* __restrict__ feat) {
  int bc = blockIdx.x;            // 0..31 over (b,c)
  int b = bc >> 4, c = bc & 15;
  __shared__ float xs[64 * 64];
  __shared__ float yr[64 * 33];
  __shared__ float yi[64 * 33];
  const float* xp = x + (size_t)bc * 4096;
  for (int i = threadIdx.x; i < 4096; i += 256) xs[i] = xp[i];
  __syncthreads();
  const float STEP = 6.283185307179586f / 64.0f;
  // stage 1: rfft along columns (s -> v)
  for (int i = threadIdx.x; i < 64 * 33; i += 256) {
    int r = i / 33, v = i % 33;
    float re = 0.f, im = 0.f;
    for (int s = 0; s < 64; ++s) {
      float sn, cs;
      __sincosf(STEP * (float)((v * s) & 63), &sn, &cs);
      float xv = xs[r * 64 + s];
      re += xv * cs;
      im -= xv * sn;
    }
    yr[i] = re; yi[i] = im;
  }
  __syncthreads();
  // stage 2: fft along rows (r -> u), ortho scale, shift + reorder, write feat
  for (int n = threadIdx.x; n < NTOK; n += 256) {
    int p = n / 33, v = n % 33;
    int u = (row_order_dev(p) + 32) & 63;   // fftshift then reorder
    float re = 0.f, im = 0.f;
    for (int r = 0; r < 64; ++r) {
      float sn, cs;
      __sincosf(STEP * (float)((u * r) & 63), &sn, &cs);
      float a = yr[r * 33 + v], bb = yi[r * 33 + v];
      re += a * cs + bb * sn;
      im += bb * cs - a * sn;
    }
    re *= (1.0f / 64.0f);
    im *= (1.0f / 64.0f);
    float* fp = feat + ((size_t)b * NTOK + n) * 32;
    fp[c]      = re;   // magnitude (real part)
    fp[16 + c] = im;   // phase (imag part)
  }
}

// ---------------------------------------------------------------------------
// Embedding: t[b,0,:] = cond*dc_w+dc_b ; t[b,n,:] = feat[b,n-1,:] @ init_w.T + init_b
// ---------------------------------------------------------------------------
__global__ void embed_kernel(const float* __restrict__ feat, const float* __restrict__ cond,
                             const float* __restrict__ init_w, const float* __restrict__ init_b,
                             const float* __restrict__ dc_w, const float* __restrict__ dc_b,
                             float* __restrict__ t) {
  size_t idx = (size_t)blockIdx.x * 256 + threadIdx.x;
  if (idx >= (size_t)MROWS * DMODEL) return;
  int d = (int)(idx & (DMODEL - 1));
  int row = (int)(idx >> 10);
  int b = row / NTOK, n = row % NTOK;
  float v;
  if (n == 0) {
    v = cond[b] * dc_w[d] + dc_b[d];
  } else {
    const float* f = feat + ((size_t)b * NTOK + (n - 1)) * 32;
    const float* w = init_w + (size_t)d * 32;
    v = init_b[d];
    #pragma unroll
    for (int k = 0; k < 32; ++k) v += f[k] * w[k];
  }
  t[idx] = v;
}

// ---------------------------------------------------------------------------
// LayerNorm: f32 in -> bf16 out, one block (256 thr) per row of 1024
// ---------------------------------------------------------------------------
__global__ void layernorm_kernel(const float* __restrict__ X, const float* __restrict__ g,
                                 const float* __restrict__ bta, __bf16* __restrict__ Y) {
  int row = blockIdx.x;
  const float* xp = X + (size_t)row * DMODEL;
  __shared__ float red[256];
  float s = 0.f;
  for (int i = threadIdx.x; i < DMODEL; i += 256) s += xp[i];
  red[threadIdx.x] = s; __syncthreads();
  for (int st = 128; st > 0; st >>= 1) {
    if (threadIdx.x < st) red[threadIdx.x] += red[threadIdx.x + st];
    __syncthreads();
  }
  float mean = red[0] * (1.0f / DMODEL);
  __syncthreads();
  float s2 = 0.f;
  for (int i = threadIdx.x; i < DMODEL; i += 256) { float dd = xp[i] - mean; s2 += dd * dd; }
  red[threadIdx.x] = s2; __syncthreads();
  for (int st = 128; st > 0; st >>= 1) {
    if (threadIdx.x < st) red[threadIdx.x] += red[threadIdx.x + st];
    __syncthreads();
  }
  float inv = rsqrtf(red[0] * (1.0f / DMODEL) + 1e-5f);
  for (int i = threadIdx.x; i < DMODEL; i += 256)
    Y[(size_t)row * DMODEL + i] = (__bf16)((xp[i] - mean) * inv * g[i] + bta[i]);
}

// ---------------------------------------------------------------------------
// WMMA GEMM: C[M,Nc] = A[M,K](bf16) * W[Nc,K](bf16)^T + bias, fused epilogues.
// Block tile 128x64, BK=32, 256 threads = 8 waves (4x2), wave tile 32x32.
// Tiles staged with CDNA5 async global->LDS DMA, double-buffered so HBM/L2
// traffic for tile k+1 overlaps WMMA on tile k (3 async b128 ops per thread
// per tile; async loads retire in order, so s_wait_asynccnt 3 == current
// tile resident while the next tile's 3 ops remain in flight).
// mode 0: outf = v + bias                 (ldc)
// mode 1: outf = v + bias + res           (ldc, residual in-place ok)
// mode 2: outb = bf16(gelu(v + bias))     (ldc)
// mode 3: head scatter: outf[((b*16+ch)*1024+code)*NTOK + tok]
// ---------------------------------------------------------------------------
__global__ __launch_bounds__(256) void gemm_wmma_kernel(
    const __bf16* __restrict__ A, int lda,
    const __bf16* __restrict__ W, int ldw,
    const float* __restrict__ bias,
    float* __restrict__ outf, __bf16* __restrict__ outb,
    const float* __restrict__ res,
    int K, int ldc, int mode)
{
  __shared__ __bf16 Abuf[2][128 * 40];
  __shared__ __bf16 Bbuf[2][64 * 40];
  int tid = threadIdx.x;
  int rowA0 = blockIdx.y * 128;
  int colB0 = blockIdx.x * 64;
  int wave = tid >> 5, lane = tid & 31;
  int wm = wave >> 1, wn = wave & 1;

  // per-thread staging chunks (b128 = 8 bf16)
  int ra0 = tid >> 2,          ca0 = (tid & 3) << 3;          // A chunk 0
  int ra1 = (tid + 256) >> 2,  ca1 = ((tid + 256) & 3) << 3;  // A chunk 1
  int rb  = tid >> 2,          cb  = (tid & 3) << 3;          // B chunk

  v8f acc[2][2];
  #pragma unroll
  for (int mt = 0; mt < 2; ++mt)
    #pragma unroll
    for (int nt = 0; nt < 2; ++nt)
      #pragma unroll
      for (int i = 0; i < 8; ++i) acc[mt][nt][i] = 0.0f;

  int ko[8];
  #pragma unroll
  for (int v = 0; v < 8; ++v) ko[v] = akoff(v, lane);

  int nk = K >> 5;

  // prologue: stage tile 0 into buffer 0
  async_copy_b128((unsigned)(size_t)&Abuf[0][ra0 * 40 + ca0],
                  A + (size_t)(rowA0 + ra0) * lda + ca0);
  async_copy_b128((unsigned)(size_t)&Abuf[0][ra1 * 40 + ca1],
                  A + (size_t)(rowA0 + ra1) * lda + ca1);
  async_copy_b128((unsigned)(size_t)&Bbuf[0][rb * 40 + cb],
                  W + (size_t)(colB0 + rb) * ldw + cb);

  for (int it = 0; it < nk; ++it) {
    int cur = it & 1;
    if (it + 1 < nk) {
      int nxt = cur ^ 1;
      int kb = (it + 1) << 5;
      async_copy_b128((unsigned)(size_t)&Abuf[nxt][ra0 * 40 + ca0],
                      A + (size_t)(rowA0 + ra0) * lda + kb + ca0);
      async_copy_b128((unsigned)(size_t)&Abuf[nxt][ra1 * 40 + ca1],
                      A + (size_t)(rowA0 + ra1) * lda + kb + ca1);
      async_copy_b128((unsigned)(size_t)&Bbuf[nxt][rb * 40 + cb],
                      W + (size_t)(colB0 + rb) * ldw + kb + cb);
      wait_async_le3();   // current tile resident; next tile still in flight
    } else {
      wait_async_0();
    }
    __syncthreads();      // all waves' portions of current tile visible

    Frag af[2], bfr[2];
    #pragma unroll
    for (int mt = 0; mt < 2; ++mt) {
      const __bf16* base = Abuf[cur] + (wm * 32 + mt * 16 + (lane & 15)) * 40;
      #pragma unroll
      for (int v = 0; v < 8; ++v) af[mt].u[v] = *(const unsigned int*)(base + ko[v]);
    }
    #pragma unroll
    for (int nt = 0; nt < 2; ++nt) {
      const __bf16* base = Bbuf[cur] + (wn * 32 + nt * 16 + (lane & 15)) * 40;
      #pragma unroll
      for (int v = 0; v < 8; ++v) bfr[nt].u[v] = *(const unsigned int*)(base + ko[v]);
    }
    #pragma unroll
    for (int mt = 0; mt < 2; ++mt)
      #pragma unroll
      for (int nt = 0; nt < 2; ++nt)
        acc[mt][nt] = wmma_bf16(af[mt], bfr[nt], acc[mt][nt]);

    __syncthreads();      // all waves done reading cur before it is re-staged
  }

  #pragma unroll
  for (int mt = 0; mt < 2; ++mt)
    #pragma unroll
    for (int nt = 0; nt < 2; ++nt)
      #pragma unroll
      for (int i = 0; i < 8; ++i) {
        int m = rowA0 + wm * 32 + mt * 16 + ((lane >> 4) << 3) + i;
        int n = colB0 + wn * 32 + nt * 16 + (lane & 15);
        float val = acc[mt][nt][i] + bias[n];
        if (mode == 0) {
          outf[(size_t)m * ldc + n] = val;
        } else if (mode == 1) {
          outf[(size_t)m * ldc + n] = val + res[(size_t)m * ldc + n];
        } else if (mode == 2) {
          float gv = 0.5f * val * (1.0f + erff(val * 0.70710678118654752f));
          outb[(size_t)m * ldc + n] = (__bf16)gv;
        } else {
          int bb = m / NTOK, tok = m % NTOK;
          int code = n >> 4, ch = n & 15;
          outf[(((size_t)bb * 16 + ch) * 1024 + code) * NTOK + tok] = val;
        }
      }
}

// ---------------------------------------------------------------------------
// RoPE + split: qkv f32 (B,N,3,H,hd) -> q,k bf16 (B,H,N,hd), vT bf16 (B,H,hd,N)
// q = q + rope(q), k = k + rope(k)
// ---------------------------------------------------------------------------
__global__ void rope_split_kernel(const float* __restrict__ qkv,
                                  __bf16* __restrict__ Qo, __bf16* __restrict__ Ko,
                                  __bf16* __restrict__ VTo) {
  int d = threadIdx.x;            // 0..127
  int n = blockIdx.x;             // token
  int bh = blockIdx.y;            // b*H + h
  int b = bh >> 3, h = bh & 7;
  const float* base = qkv + ((size_t)b * NTOK + n) * (3 * DMODEL) + h * HDIM;
  int i = d & 63;
  float div = __expf(-0.14391156831212787f * (float)i);  // exp(-ln(1e4)/64 * i)
  float sn, cs;
  __sincosf((float)n * div, &sn, &cs);
  int dr = (d < 64) ? d + 64 : d - 64;
  float sgn = (d < 64) ? -1.0f : 1.0f;

  float qv = base[0 * DMODEL + d];
  float qr = sgn * base[0 * DMODEL + dr];
  float qn = qv + (qv * cs + qr * sn);

  float kv = base[1 * DMODEL + d];
  float kr = sgn * base[1 * DMODEL + dr];
  float kn = kv + (kv * cs + kr * sn);

  float vv = base[2 * DMODEL + d];

  size_t rowoff = ((size_t)bh * NTOK + n) * HDIM + d;
  Qo[rowoff] = (__bf16)qn;
  Ko[rowoff] = (__bf16)kn;
  VTo[((size_t)bh * HDIM + d) * NTOK + n] = (__bf16)vv;   // transposed for PV WMMA
}

// ---------------------------------------------------------------------------
// Flash attention (causal), one wave per 16-row Q tile.
// Q,K: (B,H,N,hd) bf16 ; VT: (B,H,hd,N) bf16 ; O: (B,N,DMODEL) bf16
// ---------------------------------------------------------------------------
__global__ void flash_attn_kernel(const __bf16* __restrict__ Q, const __bf16* __restrict__ Kt,
                                  const __bf16* __restrict__ VT, __bf16* __restrict__ O) {
  __shared__ __bf16 Plds[16 * 32];
  int lane = threadIdx.x;
  int lm = lane & 15, lh = lane >> 4;
  int bh = blockIdx.y;
  int b = bh >> 3, h = bh & 7;
  int q0 = blockIdx.x << 4;
  const __bf16* Qp = Q + (size_t)bh * NTOK * HDIM;
  const __bf16* Kp = Kt + (size_t)bh * NTOK * HDIM;
  const __bf16* Vp = VT + (size_t)bh * HDIM * NTOK;

  int ko[8];
  #pragma unroll
  for (int v = 0; v < 8; ++v) ko[v] = akoff(v, lane);

  Frag qf[4];
  #pragma unroll
  for (int kk = 0; kk < 4; ++kk)
    #pragma unroll
    for (int v = 0; v < 8; ++v)
      qf[kk].u[v] = *(const unsigned int*)(Qp + (size_t)(q0 + lm) * HDIM + kk * 32 + ko[v]);

  float mrow[8], lrow[8];
  v8f oacc[8];
  #pragma unroll
  for (int i = 0; i < 8; ++i) { mrow[i] = -1e30f; lrow[i] = 0.0f; }
  #pragma unroll
  for (int t = 0; t < 8; ++t)
    #pragma unroll
    for (int i = 0; i < 8; ++i) oacc[t][i] = 0.0f;

  const float scale = 0.088388347648318447f;  // 1/sqrt(128)

  for (int kb = 0; kb < q0 + 16; kb += 32) {
    v8f s0, s1;
    #pragma unroll
    for (int i = 0; i < 8; ++i) { s0[i] = 0.0f; s1[i] = 0.0f; }
    #pragma unroll
    for (int kk = 0; kk < 4; ++kk) {
      Frag b0, b1;
      #pragma unroll
      for (int v = 0; v < 8; ++v) {
        b0.u[v] = *(const unsigned int*)(Kp + (size_t)(kb + lm) * HDIM + kk * 32 + ko[v]);
        b1.u[v] = *(const unsigned int*)(Kp + (size_t)(kb + 16 + lm) * HDIM + kk * 32 + ko[v]);
      }
      s0 = wmma_bf16(qf[kk], b0, s0);
      s1 = wmma_bf16(qf[kk], b1, s1);
    }
    __syncthreads();  // protect Plds reuse
    #pragma unroll
    for (int i = 0; i < 8; ++i) {
      int qrow = q0 + lh * 8 + i;
      float a  = (kb + lm      <= qrow) ? s0[i] * scale : -1e30f;
      float c2 = (kb + 16 + lm <= qrow) ? s1[i] * scale : -1e30f;
      float mx = fmaxf(a, c2);
      mx = fmaxf(mx, __shfl_xor(mx, 1, 16));
      mx = fmaxf(mx, __shfl_xor(mx, 2, 16));
      mx = fmaxf(mx, __shfl_xor(mx, 4, 16));
      mx = fmaxf(mx, __shfl_xor(mx, 8, 16));
      float mnew = fmaxf(mrow[i], mx);
      float p0 = __expf(a - mnew), p1 = __expf(c2 - mnew);
      float rs = p0 + p1;
      rs += __shfl_xor(rs, 1, 16);
      rs += __shfl_xor(rs, 2, 16);
      rs += __shfl_xor(rs, 4, 16);
      rs += __shfl_xor(rs, 8, 16);
      float corr = __expf(mrow[i] - mnew);
      lrow[i] = lrow[i] * corr + rs;
      mrow[i] = mnew;
      #pragma unroll
      for (int t = 0; t < 8; ++t) oacc[t][i] *= corr;
      Plds[(lh * 8 + i) * 32 + lm]      = (__bf16)p0;
      Plds[(lh * 8 + i) * 32 + 16 + lm] = (__bf16)p1;
    }
    __syncthreads();
    Frag pf;
    #pragma unroll
    for (int v = 0; v < 8; ++v) pf.u[v] = *(const unsigned int*)(Plds + lm * 32 + ko[v]);
    #pragma unroll
    for (int t = 0; t < 8; ++t) {
      Frag vf;
      #pragma unroll
      for (int v = 0; v < 8; ++v)
        vf.u[v] = *(const unsigned int*)(Vp + (size_t)(t * 16 + lm) * NTOK + kb + ko[v]);
      oacc[t] = wmma_bf16(pf, vf, oacc[t]);
    }
  }

  #pragma unroll
  for (int t = 0; t < 8; ++t)
    #pragma unroll
    for (int i = 0; i < 8; ++i) {
      int m = lh * 8 + i;
      float val = oacc[t][i] / lrow[i];
      O[((size_t)b * NTOK + q0 + m) * DMODEL + h * HDIM + t * 16 + lm] = (__bf16)val;
    }
}

// ---------------------------------------------------------------------------
// Host orchestration
// ---------------------------------------------------------------------------
extern "C" void kernel_launch(void* const* d_in, const int* in_sizes, int n_in,
                              void* d_out, int out_size, void* d_ws, size_t ws_size,
                              hipStream_t stream) {
  (void)in_sizes; (void)n_in; (void)out_size; (void)ws_size;
  const float* x         = (const float*)d_in[0];
  const float* condition = (const float*)d_in[1];
  const float* init_w    = (const float*)d_in[2];
  const float* init_b    = (const float*)d_in[3];
  const float* dc_w      = (const float*)d_in[4];
  const float* dc_b      = (const float*)d_in[5];
  const float* ln1_g     = (const float*)d_in[6];
  const float* ln1_b     = (const float*)d_in[7];
  const float* qkv_w     = (const float*)d_in[8];
  const float* qkv_b     = (const float*)d_in[9];
  const float* proj_w    = (const float*)d_in[10];
  const float* proj_b    = (const float*)d_in[11];
  const float* ln2_g     = (const float*)d_in[12];
  const float* ln2_b     = (const float*)d_in[13];
  const float* ff1_w     = (const float*)d_in[14];
  const float* ff1_b     = (const float*)d_in[15];
  const float* ff2_w     = (const float*)d_in[16];
  const float* ff2_b     = (const float*)d_in[17];
  const float* mag_w     = (const float*)d_in[18];
  const float* mag_b     = (const float*)d_in[19];
  const float* phase_w   = (const float*)d_in[20];
  const float* phase_b   = (const float*)d_in[21];
  float* out = (float*)d_out;

  // bump allocator in workspace (256B aligned)
  char* ws = (char*)d_ws;
  size_t off = 0;
  auto alloc = [&](size_t bytes) -> char* {
    char* p = ws + off;
    off += (bytes + 255) & ~(size_t)255;
    return p;
  };
  float*  feat = (float*) alloc((size_t)BATCH * NTOK * 32 * 4);
  float*  t    = (float*) alloc((size_t)MROWS * DMODEL * 4);
  __bf16* h    = (__bf16*)alloc((size_t)MROWS * DMODEL * 2);
  float*  qkv  = (float*) alloc((size_t)MROWS * 3 * DMODEL * 4);
  __bf16* Qb   = (__bf16*)alloc((size_t)BATCH * NHEADS * NTOK * HDIM * 2);
  __bf16* Kb   = (__bf16*)alloc((size_t)BATCH * NHEADS * NTOK * HDIM * 2);
  __bf16* VTb  = (__bf16*)alloc((size_t)BATCH * NHEADS * HDIM * NTOK * 2);
  __bf16* attn = (__bf16*)alloc((size_t)MROWS * DMODEL * 2);
  __bf16* ff   = (__bf16*)alloc((size_t)MROWS * 4 * DMODEL * 2);
  __bf16* th   = (__bf16*)alloc((size_t)MROWS * DMODEL * 2);
  __bf16* wq   = (__bf16*)alloc((size_t)3 * DMODEL * DMODEL * 2);
  __bf16* wp   = (__bf16*)alloc((size_t)DMODEL * DMODEL * 2);
  __bf16* wf1  = (__bf16*)alloc((size_t)4 * DMODEL * DMODEL * 2);
  __bf16* wf2  = (__bf16*)alloc((size_t)4 * DMODEL * DMODEL * 2);
  __bf16* wmag = (__bf16*)alloc((size_t)16384 * 512 * 2);
  __bf16* wph  = (__bf16*)alloc((size_t)16384 * 512 * 2);

  auto conv = [&](const float* s, __bf16* d, size_t n) {
    to_bf16_kernel<<<dim3((unsigned)((n + 255) / 256)), 256, 0, stream>>>(s, d, n);
  };

  // 1) FFT features
  fft_feat_kernel<<<dim3(32), 256, 0, stream>>>(x, feat);

  // 2) Embedding + shift
  embed_kernel<<<dim3((unsigned)(((size_t)MROWS * DMODEL + 255) / 256)), 256, 0, stream>>>(
      feat, condition, init_w, init_b, dc_w, dc_b, t);

  // 3) Transformer layers
  for (int l = 0; l < NLAYERS; ++l) {
    conv(qkv_w  + (size_t)l * 3 * DMODEL * DMODEL, wq,  (size_t)3 * DMODEL * DMODEL);
    conv(proj_w + (size_t)l * DMODEL * DMODEL,     wp,  (size_t)DMODEL * DMODEL);
    conv(ff1_w  + (size_t)l * 4 * DMODEL * DMODEL, wf1, (size_t)4 * DMODEL * DMODEL);
    conv(ff2_w  + (size_t)l * 4 * DMODEL * DMODEL, wf2, (size_t)4 * DMODEL * DMODEL);

    layernorm_kernel<<<dim3(MROWS), 256, 0, stream>>>(t, ln1_g + l * DMODEL, ln1_b + l * DMODEL, h);

    // qkv = h @ qkv_w.T + qkv_b   (M=4224, N=3072, K=1024)
    gemm_wmma_kernel<<<dim3(3 * DMODEL / 64, MROWS / 128), 256, 0, stream>>>(
        h, DMODEL, wq, DMODEL, qkv_b + (size_t)l * 3 * DMODEL,
        qkv, nullptr, nullptr, DMODEL, 3 * DMODEL, 0);

    rope_split_kernel<<<dim3(NTOK, BATCH * NHEADS), HDIM, 0, stream>>>(qkv, Qb, Kb, VTb);

    flash_attn_kernel<<<dim3(NTOK / 16, BATCH * NHEADS), 32, 0, stream>>>(Qb, Kb, VTb, attn);

    // t = t + attn @ proj_w.T + proj_b  (in-place residual)
    gemm_wmma_kernel<<<dim3(DMODEL / 64, MROWS / 128), 256, 0, stream>>>(
        attn, DMODEL, wp, DMODEL, proj_b + (size_t)l * DMODEL,
        t, nullptr, t, DMODEL, DMODEL, 1);

    layernorm_kernel<<<dim3(MROWS), 256, 0, stream>>>(t, ln2_g + l * DMODEL, ln2_b + l * DMODEL, h);

    // ff = gelu(h @ ff1_w.T + ff1_b)  (N=4096)
    gemm_wmma_kernel<<<dim3(4 * DMODEL / 64, MROWS / 128), 256, 0, stream>>>(
        h, DMODEL, wf1, DMODEL, ff1_b + (size_t)l * 4 * DMODEL,
        nullptr, ff, nullptr, DMODEL, 4 * DMODEL, 2);

    // t = t + ff @ ff2_w.T + ff2_b   (K=4096)
    gemm_wmma_kernel<<<dim3(DMODEL / 64, MROWS / 128), 256, 0, stream>>>(
        ff, 4 * DMODEL, wf2, 4 * DMODEL, ff2_b + (size_t)l * DMODEL,
        t, nullptr, t, 4 * DMODEL, DMODEL, 1);
  }

  // 4) Heads: split t into halves of 512, two big GEMMs with transpose scatter
  conv(t, (__bf16*)th, (size_t)MROWS * DMODEL);   // reuse as bf16 activations
  conv(mag_w,   wmag, (size_t)16384 * 512);
  conv(phase_w, wph,  (size_t)16384 * 512);

  gemm_wmma_kernel<<<dim3(16384 / 64, MROWS / 128), 256, 0, stream>>>(
      th, DMODEL, wmag, 512, mag_b, out, nullptr, nullptr, 512, 0, 3);

  gemm_wmma_kernel<<<dim3(16384 / 64, MROWS / 128), 256, 0, stream>>>(
      th + 512, DMODEL, wph, 512, phase_b, out + HEADSEG, nullptr, nullptr, 512, 0, 3);
}
